// Evolution_79740362817945
// MI455X (gfx1250) — matrix operand order repaired and checked by reading, not compiled, not run on hardware
//
#include <hip/hip_runtime.h>
#include <cstddef>

typedef __attribute__((ext_vector_type(8)))  float  v8f;
typedef __attribute__((ext_vector_type(8)))  __bf16 v8bf;
typedef __attribute__((ext_vector_type(16))) __bf16 v16bf;

#define T_STEPS 32
#define N_ROWS  4096
#define D       512    // DR2
#define KDIM    1024   // 2*DR2
#define M_BLK   16
#define THREADS 256    // 8 waves (wave32)
#define EXP_HALF 0.6065306597126334f

// ---------------------------------------------------------------------------
// One-time prep: transpose w1 [K=1024, D=512] f32 -> w1t [D=512][K=1024] bf16
// so that a B-matrix WMMA fragment (16 consecutive K of one column) is a
// contiguous 32-byte load per lane.
// ---------------------------------------------------------------------------
__global__ __launch_bounds__(THREADS)
void prep_w1_kernel(const float* __restrict__ w1, unsigned short* __restrict__ w1t_raw) {
    __bf16* w1t = (__bf16*)w1t_raw;
    int idx = blockIdx.x * THREADS + threadIdx.x;
    if (idx < D * KDIM) {
        int n = idx >> 10;           // column of w1
        int k = idx & (KDIM - 1);    // row of w1
        w1t[idx] = (__bf16)w1[(size_t)k * D + n];
    }
}

// ---------------------------------------------------------------------------
// One evolution step. Each workgroup owns M_BLK=16 full rows:
//   z = [h | s_t] @ w1  via v_wmma_f32_16x16x32_bf16 (f32 accumulate)
//   cand = sigmoid(z*thr + h*(1-thr)) * exp(-1/2)
//   delta = mean_row |cand - h| ; g = sigmoid(gl + gs*delta)
//   h_new = g*h + (1-g)*cand
// Writes states[t], diff[t-1] (t>0), h (in place), h_final (t==T-1).
// ---------------------------------------------------------------------------
__global__ __launch_bounds__(THREADS)
void evolve_step_kernel(const float* __restrict__ s_all,          // [T,N,D]
                        const float* __restrict__ thr_all,        // [T,N,1]
                        const unsigned short* __restrict__ w1t_raw, // [D][K] bf16
                        float* __restrict__ h,                    // [N,D] (in/out)
                        float* __restrict__ states,               // [T,N,D]
                        float* __restrict__ hfinal,               // [N,D]
                        float* __restrict__ diff,                 // [T-1,N,D]
                        const float* __restrict__ gl_p,
                        const float* __restrict__ gs_p,
                        int t) {
    __shared__ __bf16 sA[M_BLK * KDIM];   // 32 KB: bf16 [h | s_t] tile
    __shared__ float  sH[M_BLK * D];      // 32 KB: f32 h tile (epilogue)
    __shared__ float  sThr[M_BLK];
    __shared__ float  sPart[8][M_BLK];    // per-wave row partial sums
    __shared__ float  sRow[M_BLK];

    const __bf16* w1t = (const __bf16*)w1t_raw;
    const int tid     = threadIdx.x;
    const int rowBase = blockIdx.x * M_BLK;
    const float* s_t  = s_all + (size_t)t * N_ROWS * D;

    // ---- stage A tile as bf16 (vectorized: 8 elements / thread / iter) ----
    for (int c = tid; c < (M_BLK * KDIM) / 8; c += THREADS) {
        int idx = c * 8;
        int m = idx >> 10;            // row within tile
        int k = idx & (KDIM - 1);     // K position
        const float* src = (k < D) ? (h   + (size_t)(rowBase + m) * D + k)
                                   : (s_t + (size_t)(rowBase + m) * D + (k - D));
        float4 f0 = *(const float4*)(src);
        float4 f1 = *(const float4*)(src + 4);
        v8bf b;
        b[0] = (__bf16)f0.x; b[1] = (__bf16)f0.y; b[2] = (__bf16)f0.z; b[3] = (__bf16)f0.w;
        b[4] = (__bf16)f1.x; b[5] = (__bf16)f1.y; b[6] = (__bf16)f1.z; b[7] = (__bf16)f1.w;
        *(v8bf*)(&sA[idx]) = b;       // ds_store_b128
    }
    // ---- stage f32 h tile for the epilogue ----
    for (int c = tid; c < (M_BLK * D) / 4; c += THREADS) {
        int idx = c * 4;
        int m = idx >> 9;
        int k = idx & (D - 1);
        *(float4*)(&sH[idx]) = *(const float4*)(h + (size_t)(rowBase + m) * D + k);
    }
    if (tid < M_BLK) sThr[tid] = thr_all[(size_t)t * N_ROWS + rowBase + tid];
    __syncthreads();

    const int wave = tid >> 5;     // 0..7 -> columns [wave*64, wave*64+63]
    const int lane = tid & 31;
    const int lrow = lane & 15;    // A: M index / B: column-in-tile / D: N index
    const int half = lane >> 4;    // K-half selector for fragments

    v8f acc[4] = {};               // 4 N-tiles of 16 columns per wave

    for (int kb = 0; kb < KDIM; kb += 32) {
        // A fragment: lane holds K = kb + 16*half + {0..15} (contiguous, matches B)
        const __bf16* ap = &sA[lrow * KDIM + kb + half * 16];
        v8bf alo = *(const v8bf*)(ap);        // ds_load_b128
        v8bf ahi = *(const v8bf*)(ap + 8);    // ds_load_b128
        v16bf afrag;
#pragma unroll
        for (int i = 0; i < 8; ++i) { afrag[i] = alo[i]; afrag[i + 8] = ahi[i]; }

#pragma unroll
        for (int i = 0; i < 4; ++i) {
            int ncol = (wave * 4 + i) * 16 + lrow;
            // B fragment: 16 consecutive K of column ncol (contiguous in w1t)
            const __bf16* bp = w1t + (size_t)ncol * KDIM + kb + half * 16;
            v16bf bfrag = *(const v16bf*)bp;  // 2x global_load_b128 (L2-resident)
            acc[i] = __builtin_amdgcn_wmma_f32_16x16x32_bf16(
                false, afrag, false, bfrag, (short)0, acc[i], false, false);
        }
    }

    // ---------------- fused epilogue ----------------
    const float gl = gl_p[0];
    const float gs = gs_p[0];

    float candv[4][8];
    float hv[4][8];
    float psum[8];
#pragma unroll
    for (int r = 0; r < 8; ++r) psum[r] = 0.0f;

#pragma unroll
    for (int i = 0; i < 4; ++i) {
        int col = (wave * 4 + i) * 16 + lrow;
#pragma unroll
        for (int r = 0; r < 8; ++r) {
            int M = r + half * 8;                 // C/D layout: lanes 16-31 hold M+8
            float hval = sH[M * D + col];
            float thr  = sThr[M];
            float a    = acc[i][r] * thr + hval * (1.0f - thr);
            float cv   = EXP_HALF / (1.0f + __expf(-a));
            candv[i][r] = cv;
            hv[i][r]    = hval;
            psum[r]    += fabsf(cv - hval);
        }
    }
    // butterfly reduce across the 16 lanes sharing each row (stays within halves)
#pragma unroll
    for (int r = 0; r < 8; ++r) {
        float v = psum[r];
        v += __shfl_xor(v, 1, 32);
        v += __shfl_xor(v, 2, 32);
        v += __shfl_xor(v, 4, 32);
        v += __shfl_xor(v, 8, 32);
        psum[r] = v;
    }
    if (lrow == 0) {
#pragma unroll
        for (int r = 0; r < 8; ++r) sPart[wave][r + half * 8] = psum[r];
    }
    __syncthreads();
    if (tid < M_BLK) {
        float s = 0.0f;
#pragma unroll
        for (int w = 0; w < 8; ++w) s += sPart[w][tid];
        sRow[tid] = s;
    }
    __syncthreads();

    float grow[8];
#pragma unroll
    for (int r = 0; r < 8; ++r) {
        float delta = sRow[r + half * 8] * (1.0f / (float)D);
        grow[r] = 1.0f / (1.0f + __expf(-(gl + gs * delta)));
    }

    float* st_t = states + (size_t)t * N_ROWS * D;
#pragma unroll
    for (int i = 0; i < 4; ++i) {
        int col = (wave * 4 + i) * 16 + lrow;
#pragma unroll
        for (int r = 0; r < 8; ++r) {
            int M = r + half * 8;
            size_t off = (size_t)(rowBase + M) * D + col;
            float g  = grow[r];
            float hn = g * hv[i][r] + (1.0f - g) * candv[i][r];
            st_t[off] = hn;
            h[off]    = hn;                         // in-place state update (rows are WG-exclusive)
            if (t > 0)           diff[(size_t)(t - 1) * N_ROWS * D + off] = hn - hv[i][r];
            if (t == T_STEPS - 1) hfinal[off] = hn;
        }
    }
}

// ---------------------------------------------------------------------------
extern "C" void kernel_launch(void* const* d_in, const int* in_sizes, int n_in,
                              void* d_out, int out_size, void* d_ws, size_t ws_size,
                              hipStream_t stream) {
    const float* s_all   = (const float*)d_in[0];  // [T,N,D]
    const float* thr_all = (const float*)d_in[1];  // [T,N,1]
    const float* h0      = (const float*)d_in[2];  // [N,D]
    const float* w1      = (const float*)d_in[3];  // [K,D]
    const float* gl      = (const float*)d_in[4];  // [1]
    const float* gs      = (const float*)d_in[5];  // [1]

    // Workspace: [0, 1MB) bf16 w1^T ; [1MB, 9MB) f32 running state h
    unsigned short* w1t = (unsigned short*)d_ws;
    float* h = (float*)((char*)d_ws + (size_t)D * KDIM * sizeof(unsigned short));

    float* out    = (float*)d_out;
    float* states = out;                                       // [T,N,D]
    float* hfinal = out + (size_t)T_STEPS * N_ROWS * D;        // [N,D]
    float* diff   = hfinal + (size_t)N_ROWS * D;               // [T-1,N,D]

    hipMemcpyAsync(h, h0, (size_t)N_ROWS * D * sizeof(float),
                   hipMemcpyDeviceToDevice, stream);
    prep_w1_kernel<<<(D * KDIM + THREADS - 1) / THREADS, THREADS, 0, stream>>>(w1, w1t);

    for (int t = 0; t < T_STEPS; ++t) {
        evolve_step_kernel<<<N_ROWS / M_BLK, THREADS, 0, stream>>>(
            s_all, thr_all, w1t, h, states, hfinal, diff, gl, gs, t);
    }
}
